// DGCNN_27616639713794
// MI455X (gfx1250) — compile-verified, matrix operand order checked by reading, not run on hardware
//
#include <hip/hip_runtime.h>
#include <hip/hip_bf16.h>
#include <math.h>

// ---------------------------------------------------------------------------
// DGCNN forward for MI455X (gfx1250, wave32, WMMA).
// All GEMM-shaped work lowered to v_wmma_f32_16x16x32_bf16, with operands
// pre-packed to zero-padded bf16 rows so fragments load as b128 vectors.
// ---------------------------------------------------------------------------

typedef __attribute__((ext_vector_type(16))) __bf16        v16bf;
typedef __attribute__((ext_vector_type(8)))  float         v8f;
typedef __attribute__((ext_vector_type(4)))  unsigned int  u32x4;

union frag16 { v16bf v; u32x4 q[2]; };   // 32B: one 16x32 bf16 fragment / lane

#define NPTS   2048
#define BATCH  8
#define KNN_K  20
#define NEG_INF (-3.0e38f)
#define EPS_GN  1e-5f

// ---------------------------------------------------------------------------
// Pre-pack: f32 (R x C) -> bf16 (R x CP) zero-padded rows (CP multiple of 32,
// rows 64B-aligned so WMMA fragments are two b128 loads).
// ---------------------------------------------------------------------------
__global__ void bf16_pack_kernel(const float* __restrict__ src,
                                 __bf16* __restrict__ dst,
                                 int R, int C, int CP)
{
    int e = blockIdx.x * 256 + threadIdx.x;
    if (e >= R * CP) return;
    int r = e / CP, c = e % CP;
    dst[e] = (__bf16)((c < C) ? src[(size_t)r * C + c] : 0.f);
}

__global__ void sqnorm_kernel(const float* __restrict__ src,
                              float* __restrict__ sqn, int R, int C)
{
    int r = blockIdx.x * 256 + threadIdx.x;
    if (r >= R) return;
    float s = 0.f;
    for (int c = 0; c < C; ++c) { float v = src[(size_t)r * C + c]; s += v * v; }
    sqn[r] = s;
}

// ---------------------------------------------------------------------------
// kNN: WMMA inner products (A fragments register-cached, B fragments b128
// loads from packed bf16), -dist rows bf16 in LDS, 20x wave argmax per row.
// block = 128 (4 waves), grid = (N/16, B)
// ---------------------------------------------------------------------------
template<int CK>
__global__ void __launch_bounds__(128)
knn_topk_kernel(const __bf16* __restrict__ Xbf, const float* __restrict__ sqn,
                int* __restrict__ idx_out)
{
    const int b    = blockIdx.y;
    const int n0   = blockIdx.x * 16;
    const int tid  = threadIdx.x;
    const int wave = tid >> 5, lane = tid & 31;
    const int half = lane >> 4, l16 = lane & 15;

    __shared__ __bf16 distb[16][NPTS];       // 64 KB
    __shared__ float  sqrow[16];

    if (tid < 16) sqrow[tid] = sqn[(size_t)b * NPTS + n0 + tid];
    __syncthreads();

    const u32x4* ar = (const u32x4*)(Xbf + ((size_t)b * NPTS + (n0 + l16)) * CK);
    frag16 afr[CK / 32];
    #pragma unroll
    for (int cb = 0; cb < CK / 32; ++cb) {
        afr[cb].q[0] = ar[cb * 4 + half];
        afr[cb].q[1] = ar[cb * 4 + 2 + half];
    }
    float sArr[8];
    #pragma unroll
    for (int j = 0; j < 8; ++j) sArr[j] = sqrow[j + 8 * half];
    const float sqa_lane = sqrow[l16];   (void)sqa_lane;

    for (int mt = wave; mt < NPTS / 16; mt += 4) {
        const int m0 = mt * 16;
        const u32x4* br = (const u32x4*)(Xbf + ((size_t)b * NPTS + (m0 + l16)) * CK);
        const float sqb = sqn[(size_t)b * NPTS + m0 + l16];

        v8f acc = {};
        #pragma unroll
        for (int cb = 0; cb < CK / 32; ++cb) {
            frag16 fb;
            fb.q[0] = br[cb * 4 + half];
            fb.q[1] = br[cb * 4 + 2 + half];
            acc = __builtin_amdgcn_wmma_f32_16x16x32_bf16(
                false, afr[cb].v, false, fb.v, (short)0, acc, false, false);
        }

        const int m = m0 + l16;
        #pragma unroll
        for (int j = 0; j < 8; ++j) {
            const int r = j + 8 * half;
            float nd = 2.f * acc[j] - sArr[j] - sqb;     // -dist
            if (n0 + r == m) nd = NEG_INF;               // mask self
            distb[r][m] = (__bf16)nd;
        }
    }
    __syncthreads();

    // top-20 per row: wave w owns rows 4w..4w+3; packed u32 scan (2 bf16/word)
    for (int rr = 0; rr < 4; ++rr) {
        const int r = wave * 4 + rr;
        const unsigned int* drow = (const unsigned int*)&distb[r][0];
        for (int kk = 0; kk < KNN_K; ++kk) {
            float bestv = NEG_INF; int besti = lane;
            for (int wd = lane; wd < NPTS / 2; wd += 32) {
                unsigned int pk = drow[wd];
                unsigned int u0 = pk << 16;
                unsigned int u1 = pk & 0xffff0000u;
                float v0 = __builtin_bit_cast(float, u0);
                float v1 = __builtin_bit_cast(float, u1);
                if (v0 > bestv) { bestv = v0; besti = 2 * wd; }
                if (v1 > bestv) { bestv = v1; besti = 2 * wd + 1; }
            }
            #pragma unroll
            for (int mm = 16; mm > 0; mm >>= 1) {
                float ov = __shfl_xor(bestv, mm, 32);
                int   oi = __shfl_xor(besti, mm, 32);
                if (ov > bestv || (ov == bestv && oi < besti)) { bestv = ov; besti = oi; }
            }
            if (lane == 0)
                idx_out[((size_t)b * NPTS + (n0 + r)) * KNN_K + kk] = besti;
            distb[r][besti] = (__bf16)NEG_INF;      // same value from all lanes
        }
    }
}

// ---------------------------------------------------------------------------
// EdgeConv: h = leaky(GN4(e @ W^T)).max(k).  Two passes (recompute GEMM):
//   PASS 0: GroupNorm sum/sumsq per (batch, group) via global f32 atomics
//   PASS 1: normalize + leaky, running max over k kept in accumulator VGPRs
// Weights pre-packed bf16, row stride CK.  block = 128, grid = (N/16, B)
// ---------------------------------------------------------------------------
template<int CIN, int COUT, int PASS>
__global__ void __launch_bounds__(128)
edgeconv_kernel(const float* __restrict__ X, const int* __restrict__ knn_idx,
                const __bf16* __restrict__ Wbf, const float* __restrict__ GW,
                const float* __restrict__ GB, float* __restrict__ stats,
                float* __restrict__ Xout)
{
    constexpr int CE   = 2 * CIN;               // edge-feature channels
    constexpr int CK   = (CE + 31) & ~31;       // padded K-dim
    constexpr int NOT_ = COUT / 64;             // output tiles per wave
    constexpr int GS   = COUT / 4;              // group size (groups = 4)
    const int b    = blockIdx.y;
    const int n0   = blockIdx.x * 16;
    const int tid  = threadIdx.x;
    const int wave = tid >> 5, lane = tid & 31;
    const int half = lane >> 4, l16 = lane & 15;

    __shared__ float  ctr[16][CIN];
    __shared__ alignas(16) __bf16 Ae[16][CK];
    __shared__ int    nbr[16];

    for (int e = tid; e < 16 * CIN; e += 128) {
        int r = e / CIN, c = e % CIN;
        ctr[r][c] = X[((size_t)b * NPTS + (n0 + r)) * CIN + c];
    }
    if (CK > CE) {
        for (int e = tid; e < 16 * (CK - CE); e += 128) {
            int r = e / (CK - CE), c = CE + e % (CK - CE);
            Ae[r][c] = (__bf16)0.f;
        }
    }

    float mu[NOT_], rs[NOT_], gwv[NOT_], gbv[NOT_];
    v8f   mx[NOT_];
    float s0[NOT_], s1[NOT_];
    #pragma unroll
    for (int ot = 0; ot < NOT_; ++ot) {
        s0[ot] = 0.f; s1[ot] = 0.f;
        #pragma unroll
        for (int j = 0; j < 8; ++j) mx[ot][j] = NEG_INF;
        if (PASS == 1) {
            int o0 = (wave + 4 * ot) * 16;
            int g  = o0 / GS;
            float cnt = (float)NPTS * KNN_K * GS;
            float sm  = stats[(b * 4 + g) * 2 + 0];
            float sq  = stats[(b * 4 + g) * 2 + 1];
            float m   = sm / cnt;
            float var = sq / cnt - m * m;
            mu[ot] = m; rs[ot] = rsqrtf(var + EPS_GN);
            int o = o0 + l16;
            gwv[ot] = GW[o]; gbv[ot] = GB[o];
        }
    }

    for (int k = 0; k < KNN_K; ++k) {
        __syncthreads();
        if (tid < 16)
            nbr[tid] = knn_idx[((size_t)b * NPTS + (n0 + tid)) * KNN_K + k];
        __syncthreads();
        // edge-feature tile e = [ctr, nbr - ctr] (bf16)
        for (int e = tid; e < 16 * CIN; e += 128) {
            int r = e / CIN, c = e % CIN;
            float cv = ctr[r][c];
            float nv = X[((size_t)b * NPTS + nbr[r]) * CIN + c];
            Ae[r][c]       = (__bf16)cv;
            Ae[r][CIN + c] = (__bf16)(nv - cv);
        }
        __syncthreads();

        const u32x4* arow = (const u32x4*)&Ae[l16][0];
        #pragma unroll
        for (int ot = 0; ot < NOT_; ++ot) {
            int o0 = (wave + 4 * ot) * 16;
            const u32x4* wr = (const u32x4*)(Wbf + (size_t)(o0 + l16) * CK);
            __builtin_prefetch(wr, 0, 1);
            v8f acc = {};
            #pragma unroll
            for (int cb = 0; cb < CK / 32; ++cb) {
                frag16 fa, fb;
                fa.q[0] = arow[cb * 4 + half];
                fa.q[1] = arow[cb * 4 + 2 + half];
                fb.q[0] = wr[cb * 4 + half];
                fb.q[1] = wr[cb * 4 + 2 + half];
                acc = __builtin_amdgcn_wmma_f32_16x16x32_bf16(
                    false, fa.v, false, fb.v, (short)0, acc, false, false);
            }
            if (PASS == 0) {
                #pragma unroll
                for (int j = 0; j < 8; ++j) { s0[ot] += acc[j]; s1[ot] += acc[j] * acc[j]; }
            } else {
                #pragma unroll
                for (int j = 0; j < 8; ++j) {
                    float h = (acc[j] - mu[ot]) * rs[ot] * gwv[ot] + gbv[ot];
                    h = h > 0.f ? h : 0.2f * h;
                    mx[ot][j] = fmaxf(mx[ot][j], h);
                }
            }
        }
    }

    if (PASS == 0) {
        #pragma unroll
        for (int ot = 0; ot < NOT_; ++ot) {
            float sv = s0[ot], qv = s1[ot];
            #pragma unroll
            for (int mm = 16; mm > 0; mm >>= 1) {
                sv += __shfl_xor(sv, mm, 32);
                qv += __shfl_xor(qv, mm, 32);
            }
            if (lane == 0) {
                int g = ((wave + 4 * ot) * 16) / GS;
                atomicAdd(&stats[(b * 4 + g) * 2 + 0], sv);
                atomicAdd(&stats[(b * 4 + g) * 2 + 1], qv);
            }
        }
    } else {
        #pragma unroll
        for (int ot = 0; ot < NOT_; ++ot) {
            int o = (wave + 4 * ot) * 16 + l16;
            #pragma unroll
            for (int j = 0; j < 8; ++j) {
                int n = n0 + j + 8 * half;
                Xout[((size_t)b * NPTS + n) * COUT + o] = mx[ot][j];
            }
        }
    }
}

// ---------------------------------------------------------------------------
// Final projection: leaky(GN16(cat(x1..x4) @ wm^T)); per-tile max -> partials
// block = 256 (8 waves), grid = (N/16, B).  wm pre-packed bf16 (1024 x 512).
// ---------------------------------------------------------------------------
template<int PASS>
__global__ void __launch_bounds__(256)
proj_kernel(const float* __restrict__ x1, const float* __restrict__ x2,
            const float* __restrict__ x3, const float* __restrict__ x4,
            const __bf16* __restrict__ WMb, const float* __restrict__ GMW,
            const float* __restrict__ GMB, float* __restrict__ stats,
            float* __restrict__ partial)
{
    constexpr int CE = 512, COUT = 1024, GS = 64;   // 16 groups of 64
    const int b    = blockIdx.y;
    const int nt   = blockIdx.x;
    const int n0   = nt * 16;
    const int tid  = threadIdx.x;
    const int wave = tid >> 5, lane = tid & 31;
    const int half = lane >> 4, l16 = lane & 15;

    __shared__ alignas(16) __bf16 Ae[16][CE];       // 16 KB

    for (int e = tid; e < 16 * CE; e += 256) {
        int r = e >> 9, c = e & 511;
        size_t row = (size_t)b * NPTS + (n0 + r);
        float v;
        if      (c < 64)  v = x1[row * 64  + c];
        else if (c < 128) v = x2[row * 64  + (c - 64)];
        else if (c < 256) v = x3[row * 128 + (c - 128)];
        else              v = x4[row * 256 + (c - 256)];
        Ae[r][c] = (__bf16)v;
    }
    __syncthreads();

    const u32x4* arow = (const u32x4*)&Ae[l16][0];
    #pragma unroll
    for (int ot = 0; ot < 8; ++ot) {                // 8 waves x 8 tiles = 64
        int o0 = (wave * 8 + ot) * 16;
        int g  = o0 / GS;
        const u32x4* wr = (const u32x4*)(WMb + (size_t)(o0 + l16) * CE);
        __builtin_prefetch(wr, 0, 1);
        v8f acc = {};
        #pragma unroll
        for (int cb = 0; cb < CE / 32; ++cb) {
            frag16 fa, fb;
            fa.q[0] = arow[cb * 4 + half];
            fa.q[1] = arow[cb * 4 + 2 + half];
            fb.q[0] = wr[cb * 4 + half];
            fb.q[1] = wr[cb * 4 + 2 + half];
            acc = __builtin_amdgcn_wmma_f32_16x16x32_bf16(
                false, fa.v, false, fb.v, (short)0, acc, false, false);
        }

        if (PASS == 0) {
            float sv = 0.f, qv = 0.f;
            #pragma unroll
            for (int j = 0; j < 8; ++j) { sv += acc[j]; qv += acc[j] * acc[j]; }
            #pragma unroll
            for (int mm = 16; mm > 0; mm >>= 1) {
                sv += __shfl_xor(sv, mm, 32);
                qv += __shfl_xor(qv, mm, 32);
            }
            if (lane == 0) {
                atomicAdd(&stats[(b * 16 + g) * 2 + 0], sv);
                atomicAdd(&stats[(b * 16 + g) * 2 + 1], qv);
            }
        } else {
            float cnt = (float)NPTS * GS;
            float sm  = stats[(b * 16 + g) * 2 + 0];
            float sq  = stats[(b * 16 + g) * 2 + 1];
            float m   = sm / cnt;
            float var = sq / cnt - m * m;
            float rsv = rsqrtf(var + EPS_GN);
            int o = o0 + l16;
            float gw = GMW[o], gb = GMB[o];
            float mj = NEG_INF;
            #pragma unroll
            for (int j = 0; j < 8; ++j) {
                float h = (acc[j] - m) * rsv * gw + gb;
                h = h > 0.f ? h : 0.2f * h;
                mj = fmaxf(mj, h);
            }
            mj = fmaxf(mj, __shfl_xor(mj, 16, 32));       // combine row halves
            if (half == 0)
                partial[((size_t)(b * 128 + nt)) * COUT + o] = mj;
        }
    }
}

// ---------------------------------------------------------------------------
// Head: reduce partial maxima, then fc1+LN+leaky, fc2+LN+leaky, fc3+LN+leaky, fc4.
// single workgroup, block = 256 (8 waves, one row per wave for the LNs)
// ---------------------------------------------------------------------------
__device__ inline void ln_leaky_row(float* h, int C, const float* w,
                                    const float* bias, int lane)
{
    float s = 0.f, q = 0.f;
    for (int c = lane; c < C; c += 32) { float v = h[c]; s += v; q += v * v; }
    #pragma unroll
    for (int mm = 16; mm > 0; mm >>= 1) {
        s += __shfl_xor(s, mm, 32);
        q += __shfl_xor(q, mm, 32);
    }
    float mu = s / C, var = q / C - mu * mu, rs = rsqrtf(var + EPS_GN);
    for (int c = lane; c < C; c += 32) {
        float v = (h[c] - mu) * rs * w[c] + bias[c];
        h[c] = v > 0.f ? v : 0.2f * v;
    }
}

__global__ void __launch_bounds__(256)
head_kernel(const float* __restrict__ partial,
            const float* __restrict__ fc1w, const float* __restrict__ fc1b,
            const float* __restrict__ ln1w, const float* __restrict__ ln1b,
            const float* __restrict__ fc2w, const float* __restrict__ fc2b,
            const float* __restrict__ ln2w, const float* __restrict__ ln2b,
            const float* __restrict__ fc3w, const float* __restrict__ fc3b,
            const float* __restrict__ ln3w, const float* __restrict__ ln3b,
            const float* __restrict__ fc4w, const float* __restrict__ fc4b,
            float* __restrict__ out)
{
    __shared__ float gbuf[BATCH][1024];   // 32 KB
    __shared__ float h1[BATCH][512];      // 16 KB
    __shared__ float h2[BATCH][256];      //  8 KB
    __shared__ float h3[BATCH][64];       //  2 KB
    const int tid = threadIdx.x, wave = tid >> 5, lane = tid & 31;

    for (int e = tid; e < BATCH * 1024; e += 256) {
        int b = e >> 10, o = e & 1023;
        float m = NEG_INF;
        for (int t = 0; t < 128; ++t)
            m = fmaxf(m, partial[((size_t)(b * 128 + t)) * 1024 + o]);
        gbuf[b][o] = m;
    }
    __syncthreads();

    for (int e = tid; e < BATCH * 512; e += 256) {          // fc1
        int b = e >> 9, o = e & 511;
        float s = fc1b[o];
        for (int c = 0; c < 1024; ++c) s += gbuf[b][c] * fc1w[o * 1024 + c];
        h1[b][o] = s;
    }
    __syncthreads();
    ln_leaky_row(&h1[wave][0], 512, ln1w, ln1b, lane);
    __syncthreads();

    for (int e = tid; e < BATCH * 256; e += 256) {          // fc2
        int b = e >> 8, o = e & 255;
        float s = fc2b[o];
        for (int c = 0; c < 512; ++c) s += h1[b][c] * fc2w[o * 512 + c];
        h2[b][o] = s;
    }
    __syncthreads();
    ln_leaky_row(&h2[wave][0], 256, ln2w, ln2b, lane);
    __syncthreads();

    for (int e = tid; e < BATCH * 64; e += 256) {           // fc3
        int b = e >> 6, o = e & 63;
        float s = fc3b[o];
        for (int c = 0; c < 256; ++c) s += h2[b][c] * fc3w[o * 256 + c];
        h3[b][o] = s;
    }
    __syncthreads();
    ln_leaky_row(&h3[wave][0], 64, ln3w, ln3b, lane);
    __syncthreads();

    if (tid < BATCH * 2) {                                  // fc4
        int b = tid >> 1, o = tid & 1;
        float s = fc4b[o];
        for (int c = 0; c < 64; ++c) s += h3[b][c] * fc4w[o * 64 + c];
        out[b * 2 + o] = s;
    }
}

__global__ void zero_stats_kernel(float* stats)
{
    if (threadIdx.x < 512) stats[threadIdx.x] = 0.f;
}

// ---------------------------------------------------------------------------
extern "C" void kernel_launch(void* const* d_in, const int* in_sizes, int n_in,
                              void* d_out, int out_size, void* d_ws, size_t ws_size,
                              hipStream_t stream)
{
    const float* x    = (const float*)d_in[0];
    const float* w1   = (const float*)d_in[1];
    const float* g1w  = (const float*)d_in[2];
    const float* g1b  = (const float*)d_in[3];
    const float* w2   = (const float*)d_in[4];
    const float* g2w  = (const float*)d_in[5];
    const float* g2b  = (const float*)d_in[6];
    const float* w3   = (const float*)d_in[7];
    const float* g3w  = (const float*)d_in[8];
    const float* g3b  = (const float*)d_in[9];
    const float* w4   = (const float*)d_in[10];
    const float* g4w  = (const float*)d_in[11];
    const float* g4b  = (const float*)d_in[12];
    const float* wm   = (const float*)d_in[13];
    const float* gmw  = (const float*)d_in[14];
    const float* gmb  = (const float*)d_in[15];
    const float* fc1w = (const float*)d_in[16];
    const float* fc1b = (const float*)d_in[17];
    const float* ln1w = (const float*)d_in[18];
    const float* ln1b = (const float*)d_in[19];
    const float* fc2w = (const float*)d_in[20];
    const float* fc2b = (const float*)d_in[21];
    const float* ln2w = (const float*)d_in[22];
    const float* ln2b = (const float*)d_in[23];
    const float* fc3w = (const float*)d_in[24];
    const float* fc3b = (const float*)d_in[25];
    const float* ln3w = (const float*)d_in[26];
    const float* ln3b = (const float*)d_in[27];
    const float* fc4w = (const float*)d_in[28];
    const float* fc4b = (const float*)d_in[29];

    char* ws = (char*)d_ws;
    size_t off = 0;
    int*    idxb = (int*)(ws + off);    off += (size_t)BATCH * NPTS * KNN_K * sizeof(int);
    float*  x1   = (float*)(ws + off);  off += (size_t)BATCH * NPTS * 64  * sizeof(float);
    float*  x2   = (float*)(ws + off);  off += (size_t)BATCH * NPTS * 64  * sizeof(float);
    float*  x3   = (float*)(ws + off);  off += (size_t)BATCH * NPTS * 128 * sizeof(float);
    float*  x4   = (float*)(ws + off);  off += (size_t)BATCH * NPTS * 256 * sizeof(float);
    float*  part = (float*)(ws + off);  off += (size_t)128 * BATCH * 1024 * sizeof(float);
    float*  stats = (float*)(ws + off); off += 512 * sizeof(float);
    __bf16* xbf  = (__bf16*)(ws + off); off += (size_t)BATCH * NPTS * 128 * sizeof(__bf16);
    float*  sqn  = (float*)(ws + off);  off += (size_t)BATCH * NPTS * sizeof(float);
    __bf16* wb1  = (__bf16*)(ws + off); off += (size_t)64   * 32  * sizeof(__bf16);
    __bf16* wb2  = (__bf16*)(ws + off); off += (size_t)64   * 128 * sizeof(__bf16);
    __bf16* wb3  = (__bf16*)(ws + off); off += (size_t)128  * 128 * sizeof(__bf16);
    __bf16* wb4  = (__bf16*)(ws + off); off += (size_t)256  * 256 * sizeof(__bf16);
    __bf16* wmb  = (__bf16*)(ws + off); off += (size_t)1024 * 512 * sizeof(__bf16);
    (void)ws_size; (void)in_sizes; (void)n_in; (void)out_size;

    const int  ROWS = BATCH * NPTS;              // 16384
    const dim3 grid(NPTS / 16, BATCH);
    const float* np = (const float*)nullptr;
    float* nfp = (float*)nullptr;
    auto blks = [](int n) { return (n + 255) / 256; };

    zero_stats_kernel<<<1, 512, 0, stream>>>(stats);

    // pack all weights to padded bf16 once
    bf16_pack_kernel<<<blks(64 * 32), 256, 0, stream>>>(w1, wb1, 64, 6, 32);
    bf16_pack_kernel<<<blks(64 * 128), 256, 0, stream>>>(w2, wb2, 64, 128, 128);
    bf16_pack_kernel<<<blks(128 * 128), 256, 0, stream>>>(w3, wb3, 128, 256 / 2, 128);
    bf16_pack_kernel<<<blks(256 * 256), 256, 0, stream>>>(w4, wb4, 256, 256, 256);
    bf16_pack_kernel<<<blks(1024 * 512), 256, 0, stream>>>(wm, wmb, 1024, 512, 512);

    // ---- layer 1: 3 -> 64 ----
    bf16_pack_kernel<<<blks(ROWS * 32), 256, 0, stream>>>(x, xbf, ROWS, 3, 32);
    sqnorm_kernel<<<blks(ROWS), 256, 0, stream>>>(x, sqn, ROWS, 3);
    knn_topk_kernel<32><<<grid, 128, 0, stream>>>(xbf, sqn, idxb);
    edgeconv_kernel<3, 64, 0><<<grid, 128, 0, stream>>>(x, idxb, wb1, np, np, stats + 0, nfp);
    edgeconv_kernel<3, 64, 1><<<grid, 128, 0, stream>>>(x, idxb, wb1, g1w, g1b, stats + 0, x1);

    // ---- layer 2: 64 -> 64 ----
    bf16_pack_kernel<<<blks(ROWS * 64), 256, 0, stream>>>(x1, xbf, ROWS, 64, 64);
    sqnorm_kernel<<<blks(ROWS), 256, 0, stream>>>(x1, sqn, ROWS, 64);
    knn_topk_kernel<64><<<grid, 128, 0, stream>>>(xbf, sqn, idxb);
    edgeconv_kernel<64, 64, 0><<<grid, 128, 0, stream>>>(x1, idxb, wb2, np, np, stats + 64, nfp);
    edgeconv_kernel<64, 64, 1><<<grid, 128, 0, stream>>>(x1, idxb, wb2, g2w, g2b, stats + 64, x2);

    // ---- layer 3: 64 -> 128 ----
    bf16_pack_kernel<<<blks(ROWS * 64), 256, 0, stream>>>(x2, xbf, ROWS, 64, 64);
    sqnorm_kernel<<<blks(ROWS), 256, 0, stream>>>(x2, sqn, ROWS, 64);
    knn_topk_kernel<64><<<grid, 128, 0, stream>>>(xbf, sqn, idxb);
    edgeconv_kernel<64, 128, 0><<<grid, 128, 0, stream>>>(x2, idxb, wb3, np, np, stats + 128, nfp);
    edgeconv_kernel<64, 128, 1><<<grid, 128, 0, stream>>>(x2, idxb, wb3, g3w, g3b, stats + 128, x3);

    // ---- layer 4: 128 -> 256 ----
    bf16_pack_kernel<<<blks(ROWS * 128), 256, 0, stream>>>(x3, xbf, ROWS, 128, 128);
    sqnorm_kernel<<<blks(ROWS), 256, 0, stream>>>(x3, sqn, ROWS, 128);
    knn_topk_kernel<128><<<grid, 128, 0, stream>>>(xbf, sqn, idxb);
    edgeconv_kernel<128, 256, 0><<<grid, 128, 0, stream>>>(x3, idxb, wb4, np, np, stats + 192, nfp);
    edgeconv_kernel<128, 256, 1><<<grid, 128, 0, stream>>>(x3, idxb, wb4, g4w, g4b, stats + 192, x4);

    // projection 512 -> 1024 + GN16 + leaky + max over N (partials)
    proj_kernel<0><<<grid, 256, 0, stream>>>(x1, x2, x3, x4, wmb, np, np, stats + 256, nfp);
    proj_kernel<1><<<grid, 256, 0, stream>>>(x1, x2, x3, x4, wmb, gmw, gmb, stats + 256, part);

    // MLP head
    head_kernel<<<1, 256, 0, stream>>>(part,
        fc1w, fc1b, ln1w, ln1b, fc2w, fc2b, ln2w, ln2b,
        fc3w, fc3b, ln3w, ln3b, fc4w, fc4b, (float*)d_out);
}
